// DummyRNN_60610578481873
// MI455X (gfx1250) — matrix operand order repaired and checked by reading, not compiled, not run on hardware
//
#include <hip/hip_runtime.h>
#include <math.h>

#define HIDDEN   1024
#define T_STEPS  20480     // SEQ_NUM * SEQ_LEN, scanned serially
#define NWG      16        // persistent workgroups, one per WGP
#define ROWS_WG  64        // output rows owned per workgroup
#define NTHREADS 128       // 4 waves (wave32) -> 4 output groups of 16
#define KT_TILES 32        // 1024 / 32 K-tiles per step

typedef __attribute__((ext_vector_type(16))) __bf16 v16bf;
typedef __attribute__((ext_vector_type(8)))  float  v8f;

__device__ __forceinline__ unsigned short f2bf(float f) {
  // round-to-nearest-even float -> bf16
  unsigned u = __builtin_bit_cast(unsigned, f);
  u += 0x7FFFu + ((u >> 16) & 1u);
  return (unsigned short)(u >> 16);
}

__global__ __launch_bounds__(NTHREADS, 1)
void rnn_persistent(const float* __restrict__ xs,
                    const float* __restrict__ W_ih,
                    const float* __restrict__ b_ih,
                    const float* __restrict__ W_hh,
                    const float* __restrict__ b_hh,
                    const float* __restrict__ W_out,
                    const float* __restrict__ b_out,
                    float* __restrict__ yout,
                    unsigned* __restrict__ cnt,
                    unsigned short* __restrict__ gh0,   // h buffer, parity 0 (bf16)
                    unsigned short* __restrict__ gh1,   // h buffer, parity 1 (bf16)
                    float* __restrict__ ypart) {        // [2][NWG] y partials
  extern __shared__ char smem[];
  unsigned short* ldsW = (unsigned short*)smem;             // 128 KB swizzled W slice
  unsigned short* ldsH = (unsigned short*)(smem + 131072);  // 2 KB hidden state (bf16)
  float*          ldsY = (float*)(smem + 133120);           // per-wave y partials

  const int tid   = threadIdx.x;
  const int wg    = blockIdx.x;
  const int lane  = tid & 31;
  const int g     = tid >> 5;              // wave id == 16-row output group
  const int half  = (lane >> 4) & 1;       // lanes 0-15: rows 0..7 ; 16-31: rows 8..15
  const int mbase = wg * ROWS_WG + g * 16 + half * 8;

  // Per-lane constants for the 8 output rows this half-wave owns (redundant per lane).
  float wih[8], bias[8], wout[8];
  #pragma unroll
  for (int r = 0; r < 8; ++r) {
    const int m = mbase + r;
    wih[r]  = W_ih[m];
    bias[r] = b_ih[m] + b_hh[m];
    wout[r] = W_out[m];
  }
  const float bOut = b_out[0];

  // ---- one-time: swizzle this WGP's 64x1024 W_hh slice into LDS in WMMA-A order ----
  // record = (group g, k-tile kt, lane): 16 bf16 halves in this kernel's contiguous per-lane k order (B uses the same):
  //   lane<16 : K = kt*32 + {0..15},  lane>=16 : K = kt*32 + {16..31}
  for (int rec = tid; rec < 4 * KT_TILES * 32; rec += NTHREADS) {
    const int rg = rec >> 10;
    const int kt = (rec >> 5) & 31;
    const int rl = rec & 31;
    const int m  = wg * ROWS_WG + rg * 16 + (rl & 15);
    const int s  = (rl & 16) ? 16 : 0;
    const float* wrow = W_hh + m * HIDDEN + kt * 32 + s;
    unsigned short* dst = ldsW + rec * 16;
    #pragma unroll
    for (int j = 0; j < 16; ++j) {
      const int k = j;
      dst[j] = f2bf(wrow[k]);
    }
  }

  // Zero h(0) (all WGs write identical zeros; benign race). 128 threads * 16B = 2048B.
  ((uint4*)gh0)[tid] = make_uint4(0u, 0u, 0u, 0u);

  // ---- grid barrier 0 (monotonic counter; cnt memset to 0 by host each call) ----
  __threadfence();
  __syncthreads();
  if (tid == 0) {
    __hip_atomic_fetch_add(cnt, 1u, __ATOMIC_RELEASE, __HIP_MEMORY_SCOPE_AGENT);
    while (__hip_atomic_load(cnt, __ATOMIC_ACQUIRE, __HIP_MEMORY_SCOPE_AGENT) < (unsigned)NWG)
      __builtin_amdgcn_s_sleep(1);
  }
  __syncthreads();
  ((uint4*)ldsH)[tid] = ((const uint4*)gh0)[tid];   // stage h(0)
  __syncthreads();

  const unsigned short* aBase = ldsW + ((g * KT_TILES * 32 + lane) << 4);
  const int bsel = half * 16;

  for (int t = 0; t < T_STEPS; ++t) {
    const float x = xs[t];
    if (tid == 0 && t + 1 < T_STEPS) __builtin_prefetch(xs + t + 1, 0, 1);

    // ---- W_hh @ h via 32 v_wmma_f32_16x16x32_bf16 in 4 independent C-chains ----
    // B broadcasts h across all 16 columns -> every D column = the 16 dot products.
    // 4 interleaved accumulators cut the dependent D->C chain from 32 to 8 WMMAs.
    v8f acc0 = {0.f,0.f,0.f,0.f,0.f,0.f,0.f,0.f};
    v8f acc1 = acc0, acc2 = acc0, acc3 = acc0;
    #pragma unroll
    for (int kt = 0; kt < KT_TILES / 4; ++kt) {
      v16bf a0 = *(const v16bf*)(aBase + ((kt)               << 9));
      v16bf b0 = *(const v16bf*)(ldsH + (kt)               * 32 + bsel);
      v16bf a1 = *(const v16bf*)(aBase + ((kt + 8)          << 9));
      v16bf b1 = *(const v16bf*)(ldsH + (kt + 8)          * 32 + bsel);
      v16bf a2 = *(const v16bf*)(aBase + ((kt + 16)         << 9));
      v16bf b2 = *(const v16bf*)(ldsH + (kt + 16)         * 32 + bsel);
      v16bf a3 = *(const v16bf*)(aBase + ((kt + 24)         << 9));
      v16bf b3 = *(const v16bf*)(ldsH + (kt + 24)         * 32 + bsel);
      acc0 = __builtin_amdgcn_wmma_f32_16x16x32_bf16(false, a0, false, b0, (short)0, acc0, false, false);
      acc1 = __builtin_amdgcn_wmma_f32_16x16x32_bf16(false, a1, false, b1, (short)0, acc1, false, false);
      acc2 = __builtin_amdgcn_wmma_f32_16x16x32_bf16(false, a2, false, b2, (short)0, acc2, false, false);
      acc3 = __builtin_amdgcn_wmma_f32_16x16x32_bf16(false, a3, false, b3, (short)0, acc3, false, false);
    }
    const v8f acc = (acc0 + acc1) + (acc2 + acc3);

    // ---- elementwise: h_new = tanh(x*w_ih + bias + acc); y partial ----
    float yp = 0.f;
    unsigned short hb[8];
    #pragma unroll
    for (int r = 0; r < 8; ++r) {
      const float pre = fmaf(x, wih[r], bias[r]) + acc[r];
      const float hn  = tanhf(pre);
      yp = fmaf(wout[r], hn, yp);
      hb[r] = f2bf(hn);
    }

    unsigned short* dstH = (t & 1) ? gh0 : gh1;   // h(t+1) -> buffer (t+1)%2
    if ((lane & 15) == 0) {                        // lanes 0 and 16 publish 8 rows each
      union { unsigned short u[8]; uint4 v; } pk;
      #pragma unroll
      for (int r = 0; r < 8; ++r) pk.u[r] = hb[r];
      *(uint4*)(dstH + mbase) = pk.v;
    }
    const float ypo = __shfl_down(yp, 16, 32);     // combine halves of the group
    if (lane == 0) ldsY[g] = yp + ypo;

    // ---- per-step grid barrier (release stores, arrive, acquire spin) ----
    __threadfence();
    __syncthreads();
    if (tid == 0) {
      ypart[(t & 1) * NWG + wg] = ldsY[0] + ldsY[1] + ldsY[2] + ldsY[3];
      __hip_atomic_fetch_add(cnt, 1u, __ATOMIC_RELEASE, __HIP_MEMORY_SCOPE_AGENT);
      const unsigned target = (unsigned)NWG * (unsigned)(t + 2);
      while (__hip_atomic_load(cnt, __ATOMIC_ACQUIRE, __HIP_MEMORY_SCOPE_AGENT) < target)
        __builtin_amdgcn_s_sleep(1);
    }
    __syncthreads();

    // WG0 deterministically reduces the 16 partials -> y[t]
    if (wg == 0 && tid == 0) {
      const float* yp2 = ypart + (t & 1) * NWG;
      float y = bOut;
      #pragma unroll
      for (int i = 0; i < NWG; ++i) y += yp2[i];
      yout[t] = y;
    }

    // Stage h(t+1) into LDS for the next step.
    ((uint4*)ldsH)[tid] = ((const uint4*)dstH)[tid];
    __syncthreads();
  }
}

extern "C" void kernel_launch(void* const* d_in, const int* in_sizes, int n_in,
                              void* d_out, int out_size, void* d_ws, size_t ws_size,
                              hipStream_t stream) {
  (void)in_sizes; (void)n_in; (void)out_size; (void)ws_size;
  const float* xs    = (const float*)d_in[0];
  const float* W_ih  = (const float*)d_in[1];
  const float* b_ih  = (const float*)d_in[2];
  const float* W_hh  = (const float*)d_in[3];
  const float* b_hh  = (const float*)d_in[4];
  const float* W_out = (const float*)d_in[5];
  const float* b_out = (const float*)d_in[6];
  float* yout = (float*)d_out;

  unsigned char* ws = (unsigned char*)d_ws;
  unsigned*       cnt   = (unsigned*)ws;                       // barrier counter
  unsigned short* gh0   = (unsigned short*)(ws + 256);         // 2 KB
  unsigned short* gh1   = (unsigned short*)(ws + 256 + 2048);  // 2 KB
  float*          ypart = (float*)(ws + 256 + 4096);           // 2*NWG floats

  // Zero the barrier counter every call (graph-capture-safe memset node).
  hipMemsetAsync(ws, 0, 256, stream);

  const int SMEM = 131072 + 2048 + 64;   // W slice + h + per-wave y partials
  hipFuncSetAttribute((const void*)rnn_persistent,
                      hipFuncAttributeMaxDynamicSharedMemorySize, SMEM);

  rnn_persistent<<<dim3(NWG), dim3(NTHREADS), SMEM, stream>>>(
      xs, W_ih, b_ih, W_hh, b_hh, W_out, b_out, yout, cnt, gh0, gh1, ypart);
}
